// MistralAttention_81123342287477
// MI455X (gfx1250) — matrix operand order repaired and checked by reading, not compiled, not run on hardware
//
#include <hip/hip_runtime.h>
#include <hip/hip_bf16.h>

typedef __attribute__((ext_vector_type(16))) __bf16 v16bf;
typedef __attribute__((ext_vector_type(8)))  float  v8f;
typedef unsigned int u32x4 __attribute__((ext_vector_type(4)));
typedef int          i32x4 __attribute__((ext_vector_type(4)));
typedef int          i32x8 __attribute__((ext_vector_type(8)));

static constexpr int BB   = 2;
static constexpr int S    = 2048;
static constexpr int H    = 4096;
static constexpr int NH   = 32;
static constexpr int NKV  = 8;
static constexpr int HD   = 128;
static constexpr int KVD  = NKV * HD;     // 1024
static constexpr int ROWS = BB * S;       // 4096
static constexpr int WIN  = 512;          // SLIDING_WINDOW / 2

// ---------------------------------------------------------------------------
// Tensor Data Mover: async 2-D bf16 tile load Global -> LDS.
// Builds a D# per CDNA5 ISA ch.8: group0 = {count/lds_addr/global_addr/type},
// group1 = {data_size, tensor_dim0/1, tile_dim0/1, tensor_dim0_stride}.
// Rows beyond `rows_avail` (tensor_dim1) are zero-filled by hardware (OOB).
// All operands forced uniform (SGPR) via readfirstlane.
// ---------------------------------------------------------------------------
__device__ __forceinline__ void tdm_load_2d(uint32_t lds_off, const void* gptr,
                                            uint32_t width, uint32_t tile_rows,
                                            uint32_t rows_avail, uint32_t stride) {
  uint64_t ga = (uint64_t)(uintptr_t)gptr;
  uint32_t ga_lo = (uint32_t)__builtin_amdgcn_readfirstlane((int)(uint32_t)ga);
  uint32_t ga_hi = (uint32_t)__builtin_amdgcn_readfirstlane((int)(uint32_t)(ga >> 32));
  lds_off    = (uint32_t)__builtin_amdgcn_readfirstlane((int)lds_off);
  width      = (uint32_t)__builtin_amdgcn_readfirstlane((int)width);
  tile_rows  = (uint32_t)__builtin_amdgcn_readfirstlane((int)tile_rows);
  rows_avail = (uint32_t)__builtin_amdgcn_readfirstlane((int)rows_avail);
  stride     = (uint32_t)__builtin_amdgcn_readfirstlane((int)stride);

  u32x4 g0;
  g0[0] = 1u;                                    // count = 1 (valid user D#)
  g0[1] = lds_off;                               // lds_addr (bytes)
  g0[2] = ga_lo;                                 // global_addr[31:0]
  g0[3] = (ga_hi & 0x1FFFFFFu) | (2u << 30);     // global_addr[56:32] | type=2

  i32x8 g1;
  g1[0] = (int)(1u << 16);                       // data_size = 1 (2 bytes/elem)
  g1[1] = (int)(width << 16);                    // tensor_dim0[15:0]
  g1[2] = (int)((width >> 16) | (rows_avail << 16));   // dim0 hi | tensor_dim1 lo
  g1[3] = (int)((rows_avail >> 16) | (width << 16));   // dim1 hi | tile_dim0
  g1[4] = (int)(tile_rows & 0xFFFFu);            // tile_dim1 (tile_dim2 = 0)
  g1[5] = (int)stride;                           // tensor_dim0_stride[31:0]
  g1[6] = 0;
  g1[7] = 0;

  i32x4 g2 = (i32x4)0;
  i32x4 g3 = (i32x4)0;
#if defined(__clang_major__) && (__clang_major__ >= 23)
  i32x8 g4 = (i32x8)0;
  __builtin_amdgcn_tensor_load_to_lds(g0, g1, g2, g3, g4, 0);
#else
  __builtin_amdgcn_tensor_load_to_lds(g0, g1, g2, g3, 0);
#endif
}

__device__ __forceinline__ uint32_t lds_offset_of(const void* p) {
  return (uint32_t)(uintptr_t)p;   // LDS aperture: flat addr[31:0] == LDS byte offset
}

// ---------------------------------------------------------------------------
// Fragment loaders for V_WMMA_F32_16X16X32_BF16 (16 contiguous k per lane; callers use k = 16h + s)
// ---------------------------------------------------------------------------
__device__ __forceinline__ v16bf load_afrag(const __bf16* p) {
  v16bf a;
  const uint4* q = (const uint4*)p;
  ((uint4*)&a)[0] = q[0];   // K = off .. off+7
  ((uint4*)&a)[1] = q[1];   // K = off+8 .. off+15
  return a;
}
__device__ __forceinline__ v16bf load_bfrag(const __bf16* p) {
  v16bf b;
  const uint4* q = (const uint4*)p;
  ((uint4*)&b)[0] = q[0];
  ((uint4*)&b)[1] = q[1];
  return b;
}

// ---------------------------------------------------------------------------
// fp32 -> bf16 conversion
// ---------------------------------------------------------------------------
__global__ void cvt_bf16_kernel(const float* __restrict__ X, __bf16* __restrict__ Y, int n) {
  int i = blockIdx.x * blockDim.x + threadIdx.x;
  if (i < n) Y[i] = (__bf16)X[i];
}

__global__ void cvt_transpose_kernel(const float* __restrict__ W, __bf16* __restrict__ Wt,
                                     int K, int N) {
  int idx = blockIdx.x * blockDim.x + threadIdx.x;
  if (idx >= K * N) return;
  int k = idx % K;
  int n = idx / K;
  Wt[(size_t)n * K + k] = (__bf16)W[(size_t)k * N + n];
}

// ---------------------------------------------------------------------------
// bf16 GEMM: C[M x N] = A[M x K] * Bt[N x K]^T, fp32 accumulate via WMMA.
// Block = 256 threads = 8 waves (2M x 4N); wave tile 16x64.
// Bt tiles (256 x 32) streamed by TDM into double-buffered LDS.
// ---------------------------------------------------------------------------
template <bool F32OUT>
__global__ __launch_bounds__(256) void gemm_bt_wmma(const __bf16* __restrict__ A,
                                                    const __bf16* __restrict__ Bt,
                                                    void* __restrict__ Cout,
                                                    int M, int N, int K) {
  __shared__ __align__(16) __bf16 shB[2][256][32];

  const int tid = threadIdx.x;
  const int w   = tid >> 5;
  const int l   = tid & 31;
  const int l16 = l & 15;
  const int lh  = l >> 4;
  const int wm  = w >> 2;
  const int wn  = w & 3;
  const int mb  = blockIdx.x * 32 + wm * 16;
  const int nb0 = blockIdx.y * 256;

  const int arow = mb + l16;
  const int aoff = lh ? 16 : 0;
  const int boff = lh ? 16 : 0;

  const __bf16* Bblk = Bt + (size_t)nb0 * K;

  v8f c0 = {}, c1 = {}, c2 = {}, c3 = {};
  const int kt = K >> 5;

  if (tid < 32) {   // wave 0 drives the TDM
    tdm_load_2d(lds_offset_of(&shB[0][0][0]), Bblk, 32, 256, 256, (uint32_t)K);
    __builtin_amdgcn_s_wait_tensorcnt(0);
  }
  __syncthreads();

  int cur = 0;
  for (int kc = 0; kc < kt; ++kc) {
    const bool more = (kc + 1) < kt;
    if (more && tid < 32)   // async-stage next K-chunk while computing this one
      tdm_load_2d(lds_offset_of(&shB[cur ^ 1][0][0]), Bblk + (size_t)(kc + 1) * 32,
                  32, 256, 256, (uint32_t)K);

    const __bf16* ap = A + (size_t)arow * K + kc * 32 + aoff;
    v16bf a = load_afrag(ap);
    if (kc + 8 < kt) __builtin_prefetch(ap + 8 * 32, 0, 1);   // global_prefetch_b8

    const __bf16* bp = &shB[cur][wn * 64 + l16][boff];
    v16bf b0 = load_bfrag(bp);
    v16bf b1 = load_bfrag(bp + 16 * 32);
    v16bf b2 = load_bfrag(bp + 32 * 32);
    v16bf b3 = load_bfrag(bp + 48 * 32);

    c0 = __builtin_amdgcn_wmma_f32_16x16x32_bf16(false, a, false, b0, (short)0, c0, false, false);
    c1 = __builtin_amdgcn_wmma_f32_16x16x32_bf16(false, a, false, b1, (short)0, c1, false, false);
    c2 = __builtin_amdgcn_wmma_f32_16x16x32_bf16(false, a, false, b2, (short)0, c2, false, false);
    c3 = __builtin_amdgcn_wmma_f32_16x16x32_bf16(false, a, false, b3, (short)0, c3, false, false);

    if (more && tid < 32) __builtin_amdgcn_s_wait_tensorcnt(0);
    __syncthreads();
    cur ^= 1;
  }

  for (int r = 0; r < 8; ++r) {
    int grow = mb + r + 8 * lh;
    size_t base = (size_t)grow * N + nb0 + wn * 64 + l16;
    if (F32OUT) {
      float* o = (float*)Cout;
      o[base]      = c0[r];
      o[base + 16] = c1[r];
      o[base + 32] = c2[r];
      o[base + 48] = c3[r];
    } else {
      __bf16* o = (__bf16*)Cout;
      o[base]      = (__bf16)c0[r];
      o[base + 16] = (__bf16)c1[r];
      o[base + 32] = (__bf16)c2[r];
      o[base + 48] = (__bf16)c3[r];
    }
  }
}

// ---------------------------------------------------------------------------
// RoPE in-place on bf16 Q or K
// ---------------------------------------------------------------------------
__global__ void rope_kernel(__bf16* __restrict__ X, int nheads) {
  int idx = blockIdx.x * blockDim.x + threadIdx.x;
  int d = idx & 63;
  int h = (idx >> 6) % nheads;
  int row = idx / (64 * nheads);
  if (row >= ROWS) return;
  int pos = row % S;
  float freq = (float)pos * __powf(10000.0f, -(float)d / 64.0f);
  float cs, sn;
  __sincosf(freq, &sn, &cs);
  size_t base = (size_t)row * (nheads * HD) + h * HD + d;
  float x1 = (float)X[base];
  float x2 = (float)X[base + 64];
  X[base]      = (__bf16)(x1 * cs - x2 * sn);
  X[base + 64] = (__bf16)(x2 * cs + x1 * sn);
}

// ---------------------------------------------------------------------------
// Sliding-window flash attention, GQA, TDM-double-buffered K/V staging.
// Grid: (B*NH, S/64); block = 128 threads = 4 waves; wave w owns 16 queries.
// ---------------------------------------------------------------------------
__global__ __launch_bounds__(128) void attn_fa_kernel(const __bf16* __restrict__ Q,
                                                      const __bf16* __restrict__ Kk,
                                                      const __bf16* __restrict__ Vv,
                                                      __bf16* __restrict__ O) {
  __shared__ __align__(16) __bf16 shK[2][32][128];
  __shared__ __align__(16) __bf16 shV[2][32][128];
  __shared__ __align__(16) __bf16 shP[4][16][32];

  const int tid = threadIdx.x;
  const int w   = tid >> 5;
  const int l   = tid & 31;
  const int l16 = l & 15;
  const int lh  = l >> 4;

  const int b   = blockIdx.x >> 5;
  const int h   = blockIdx.x & 31;
  const int kvh = h >> 2;
  const int qt0 = blockIdx.y * 64;
  const int srow = qt0 + w * 16;
  const int aoff = lh ? 16 : 0;

  // Q A-fragments (4 head-dim chunks of 32)
  v16bf aq[4];
  {
    size_t qbase = (size_t)(b * S + srow + l16) * H + h * HD;
    for (int dc = 0; dc < 4; ++dc)
      aq[dc] = load_afrag(Q + qbase + dc * 32 + aoff);
  }

  v8f oacc[8] = {};
  float mrow[8], lrow[8];
  for (int r = 0; r < 8; ++r) { mrow[r] = -1e30f; lrow[r] = 0.f; }

  int kstart = qt0 - WIN;
  if (kstart < 0) kstart = 0;
  kstart &= ~31;
  const int kend = qt0 + 63;

  const size_t kvbase = (size_t)(b * S) * KVD + kvh * HD;

  // prologue: TDM-stage the first K/V tile; OOB rows (>= S) zero-filled by HW
  if (tid < 32) {
    tdm_load_2d(lds_offset_of(&shK[0][0][0]), Kk + kvbase + (size_t)kstart * KVD,
                128, 32, (uint32_t)(S - kstart), KVD);
    tdm_load_2d(lds_offset_of(&shV[0][0][0]), Vv + kvbase + (size_t)kstart * KVD,
                128, 32, (uint32_t)(S - kstart), KVD);
    __builtin_amdgcn_s_wait_tensorcnt(0);
  }
  __syncthreads();

  int cur = 0;
  for (int kt2 = kstart; kt2 <= kend; kt2 += 32) {
    const bool more = (kt2 + 32) <= kend;
    if (more && tid < 32) {   // async-stage next tile while computing this one
      int nt2 = kt2 + 32;
      tdm_load_2d(lds_offset_of(&shK[cur ^ 1][0][0]), Kk + kvbase + (size_t)nt2 * KVD,
                  128, 32, (uint32_t)(S - nt2), KVD);
      tdm_load_2d(lds_offset_of(&shV[cur ^ 1][0][0]), Vv + kvbase + (size_t)nt2 * KVD,
                  128, 32, (uint32_t)(S - nt2), KVD);
    }

    // ---- scores + online softmax for two 16-key sub-tiles ----
    for (int sub = 0; sub < 2; ++sub) {
      v16bf bk0 = load_bfrag(&shK[cur][sub * 16 + l16][0   + (lh ? 16 : 0)]);
      v16bf bk1 = load_bfrag(&shK[cur][sub * 16 + l16][32  + (lh ? 16 : 0)]);
      v16bf bk2 = load_bfrag(&shK[cur][sub * 16 + l16][64  + (lh ? 16 : 0)]);
      v16bf bk3 = load_bfrag(&shK[cur][sub * 16 + l16][96  + (lh ? 16 : 0)]);
      v8f sc = {};
      sc = __builtin_amdgcn_wmma_f32_16x16x32_bf16(false, aq[0], false, bk0, (short)0, sc, false, false);
      sc = __builtin_amdgcn_wmma_f32_16x16x32_bf16(false, aq[1], false, bk1, (short)0, sc, false, false);
      sc = __builtin_amdgcn_wmma_f32_16x16x32_bf16(false, aq[2], false, bk2, (short)0, sc, false, false);
      sc = __builtin_amdgcn_wmma_f32_16x16x32_bf16(false, aq[3], false, bk3, (short)0, sc, false, false);

      int j = kt2 + sub * 16 + l16;
      for (int r = 0; r < 8; ++r) {
        int i = srow + r + 8 * lh;
        float s = sc[r] * 0.08838834764831845f;   // 1/sqrt(128)
        bool ok = (j <= i) && (i - j <= WIN);
        s = ok ? s : -1e30f;
        float v = s;
        for (int off = 8; off; off >>= 1) v = fmaxf(v, __shfl_xor(v, off, 16));
        float mnew = fmaxf(mrow[r], v);
        float corr = __expf(mrow[r] - mnew);
        mrow[r] = mnew;
        float pe = ok ? __expf(s - mnew) : 0.f;
        float ps = pe;
        for (int off = 8; off; off >>= 1) ps += __shfl_xor(ps, off, 16);
        lrow[r] = lrow[r] * corr + ps;
        for (int nt = 0; nt < 8; ++nt) oacc[nt][r] *= corr;
        shP[w][r + 8 * lh][sub * 16 + l16] = (__bf16)pe;   // transpose via LDS
      }
    }
    __syncthreads();

    // ---- P(16x32) @ V(32x128) ----
    v16bf ap = load_afrag(&shP[w][l16][aoff]);
    for (int nt = 0; nt < 8; ++nt) {
      v16bf bv;
      int col = nt * 16 + l16;
      int kb  = lh ? 16 : 0;
      for (int j2 = 0; j2 < 16; ++j2) bv[j2] = shV[cur][kb + j2][col];
      oacc[nt] = __builtin_amdgcn_wmma_f32_16x16x32_bf16(false, ap, false, bv,
                                                         (short)0, oacc[nt], false, false);
    }

    if (more && tid < 32) __builtin_amdgcn_s_wait_tensorcnt(0);
    __syncthreads();
    cur ^= 1;
  }

  for (int r = 0; r < 8; ++r) {
    float inv = 1.0f / lrow[r];
    size_t base = (size_t)(b * S + srow + r + 8 * lh) * H + h * HD + l16;
    for (int nt = 0; nt < 8; ++nt)
      O[base + nt * 16] = (__bf16)(oacc[nt][r] * inv);
  }
}

// ---------------------------------------------------------------------------
// Host-side orchestration
// ---------------------------------------------------------------------------
extern "C" void kernel_launch(void* const* d_in, const int* in_sizes, int n_in,
                              void* d_out, int out_size, void* d_ws, size_t ws_size,
                              hipStream_t stream) {
  const float* hs = (const float*)d_in[0];
  const float* Wq = (const float*)d_in[1];
  const float* Wk = (const float*)d_in[2];
  const float* Wv = (const float*)d_in[3];
  const float* Wo = (const float*)d_in[4];
  float* out = (float*)d_out;

  char* ws = (char*)d_ws;
  size_t off = 0;
  auto alloc = [&](size_t bytes) -> size_t {
    size_t o = off;
    off += (bytes + 255) & ~(size_t)255;
    return o;
  };
  __bf16* hsb = (__bf16*)(ws + alloc((size_t)ROWS * H * 2));
  __bf16* wqT = (__bf16*)(ws + alloc((size_t)H * H * 2));
  __bf16* wkT = (__bf16*)(ws + alloc((size_t)H * KVD * 2));
  __bf16* wvT = (__bf16*)(ws + alloc((size_t)H * KVD * 2));
  __bf16* woT = (__bf16*)(ws + alloc((size_t)H * H * 2));
  __bf16* qb  = (__bf16*)(ws + alloc((size_t)ROWS * H * 2));
  __bf16* kb  = (__bf16*)(ws + alloc((size_t)ROWS * KVD * 2));
  __bf16* vb  = (__bf16*)(ws + alloc((size_t)ROWS * KVD * 2));
  __bf16* ab  = (__bf16*)(ws + alloc((size_t)ROWS * H * 2));

  cvt_bf16_kernel<<<(ROWS * H) / 256, 256, 0, stream>>>(hs, hsb, ROWS * H);
  cvt_transpose_kernel<<<(H * H) / 256, 256, 0, stream>>>(Wq, wqT, H, H);
  cvt_transpose_kernel<<<(H * KVD) / 256, 256, 0, stream>>>(Wk, wkT, H, KVD);
  cvt_transpose_kernel<<<(H * KVD) / 256, 256, 0, stream>>>(Wv, wvT, H, KVD);
  cvt_transpose_kernel<<<(H * H) / 256, 256, 0, stream>>>(Wo, woT, H, H);

  gemm_bt_wmma<false><<<dim3(ROWS / 32, H / 256), 256, 0, stream>>>(hsb, wqT, qb, ROWS, H, H);
  gemm_bt_wmma<false><<<dim3(ROWS / 32, KVD / 256), 256, 0, stream>>>(hsb, wkT, kb, ROWS, KVD, H);
  gemm_bt_wmma<false><<<dim3(ROWS / 32, KVD / 256), 256, 0, stream>>>(hsb, wvT, vb, ROWS, KVD, H);

  rope_kernel<<<(ROWS * NH * 64) / 256, 256, 0, stream>>>(qb, NH);
  rope_kernel<<<(ROWS * NKV * 64) / 256, 256, 0, stream>>>(kb, NKV);

  attn_fa_kernel<<<dim3(BB * NH, S / 64), 128, 0, stream>>>(qb, kb, vb, ab);

  gemm_bt_wmma<true><<<dim3(ROWS / 32, H / 256), 256, 0, stream>>>(ab, woT, (void*)out, ROWS, H, H);
}
